// st_attention_40819369181200
// MI455X (gfx1250) — compile-verified, matrix operand order checked by reading, not run on hardware
//
#include <hip/hip_runtime.h>

// ---------------------------------------------------------------------------
// ST-attention (graph attention over NTU skeleton) for MI455X / gfx1250.
// Shapes: B=128, C=CO=64, T=128, V=25, H=3, RC2=4, RC=8.
//
// Restructured math (P folded into per-(b,h,o) 25x25 matrix M):
//   out[b,o,t,v] = sum_h ( sum_u Y_h[b,o,t,u] * M_h[b,o,u,v]
//                          + wv_b[h,o] * srow_h[b,o,v] )
//   Y_h[b,o,t,u] = sum_c wv_h[o,c] * x[b,c,t,u]       <- bf16 WMMA GEMM
//   M_h[o,u,v]   = sum_w P_h[u,w] * s_h[o,v,w]        <- tiny, precomputed
// then train-mode BatchNorm + residual(x) + ReLU.
//
// All WMMA operands are staged pre-swizzled so every operand fetch is a pair
// of b128 loads (global for A/Wv and B/M, LDS for B/X and A/Y).
// ---------------------------------------------------------------------------

typedef __attribute__((ext_vector_type(16))) __bf16 v16bf;
typedef __attribute__((ext_vector_type(8)))  float  v8f;

#define B_   128
#define C_   64
#define CO_  64
#define T_   128
#define V_   25
#define H_   3
#define BCH  32   // batch chunk (bounds M scratch to ~12.6 MB)

// workspace byte offsets (all 1 KB aligned)
#define OFF_P      0            // H*V*V f32            (7.5 KB)
#define OFF_XMEAN  16384        // B*C*V f32            (800 KB)
#define OFF_SROW   851968       // BCH*H*CO*V f32       (600 KB)
#define OFF_STATS  1468416      // 2*64 f32 (sum,sumsq)
#define OFF_BNP    1469440      // 2*64 f32 (mean,scale)
#define OFF_WVA    1470464      // 3*4*2*512 bf16       (24 KB)  A-operand swizzled Wv
#define OFF_MWS    1495040      // BCH*H*CO*2*512 bf16  (12.6 MB) B-operand swizzled M

// Assumed 16-bit WMMA operand K packing (ISA 7.12.2): element e of the
// per-lane v16bf; lanes >=16 hold the +8 K-offset halves.
__device__ __forceinline__ int kmap(int e, int lane) {
  int kh = (lane < 16) ? 0 : 8;
  return (e < 8) ? (kh + e) : (16 + kh + (e - 8));
}
__device__ __forceinline__ unsigned short f2bf(float f) {
  union { float f; unsigned u; } x; x.f = f;
  return (unsigned short)(x.u >> 16);
}
__device__ __forceinline__ unsigned pk2(float lo, float hi) {
  return (unsigned)f2bf(lo) | ((unsigned)f2bf(hi) << 16);
}

// --------------------------- K1: Sinkhorn -----------------------------------
__global__ void k1_sinkhorn(const float* __restrict__ log_alpha, float* __restrict__ P) {
  __shared__ float la[V_ * V_];
  int h = blockIdx.x, tid = threadIdx.x;
  for (int i = tid; i < V_ * V_; i += 32) la[i] = log_alpha[h * V_ * V_ + i];
  __syncthreads();
  for (int it = 0; it < 20; ++it) {
    if (tid < V_) {                       // row logsumexp (axis=1)
      float m = -1e30f;
      for (int w = 0; w < V_; w++) m = fmaxf(m, la[tid * V_ + w]);
      float s = 0.f;
      for (int w = 0; w < V_; w++) s += expf(la[tid * V_ + w] - m);
      float lse = m + logf(s);
      for (int w = 0; w < V_; w++) la[tid * V_ + w] -= lse;
    }
    __syncthreads();
    if (tid < V_) {                       // col logsumexp (axis=0)
      float m = -1e30f;
      for (int u = 0; u < V_; u++) m = fmaxf(m, la[u * V_ + tid]);
      float s = 0.f;
      for (int u = 0; u < V_; u++) s += expf(la[u * V_ + tid] - m);
      float lse = m + logf(s);
      for (int u = 0; u < V_; u++) la[u * V_ + tid] -= lse;
    }
    __syncthreads();
  }
  for (int i = tid; i < V_ * V_; i += 32) P[h * V_ * V_ + i] = expf(la[i]);
}

// --------------------------- K2: time-mean of x -----------------------------
__global__ __launch_bounds__(128)
void k2_xmean(const float* __restrict__ x, float* __restrict__ xmean) {
  __shared__ float sv[V_];
  int bc = blockIdx.x, tid = threadIdx.x;
  if (tid < V_) sv[tid] = 0.f;
  __syncthreads();
  const float* xr = x + (size_t)bc * T_ * V_;
  for (int i = tid; i < T_ * V_; i += 128) atomicAdd(&sv[i % V_], xr[i]);
  __syncthreads();
  if (tid < V_) xmean[(size_t)bc * V_ + tid] = sv[tid] * (1.0f / (float)T_);
}

// --------------------------- K0: Wv -> A-operand bf16 -----------------------
__global__ void k0_wva(const float* __restrict__ wv_w, unsigned short* __restrict__ wvA) {
  int idx = blockIdx.x * 256 + threadIdx.x;
  if (idx >= H_ * 4 * 2 * 512) return;
  int e = idx & 15, l = (idx >> 4) & 31, kk = (idx >> 9) & 1, ot = (idx >> 10) & 3, h = idx >> 12;
  int o = ot * 16 + (l & 15);
  int c = kk * 32 + kmap(e, l);
  wvA[idx] = f2bf(wv_w[(h * CO_ + o) * C_ + c]);
}

// --------------------------- K3: scores -> M (B-operand bf16), srow ---------
__global__ __launch_bounds__(256)
void k3_scores(const float* __restrict__ xmean, const float* __restrict__ P,
               const float* __restrict__ wk_w, const float* __restrict__ wk_b,
               const float* __restrict__ wq_w, const float* __restrict__ wq_b,
               const float* __restrict__ c1_w, const float* __restrict__ c1_b,
               const float* __restrict__ c2_w, const float* __restrict__ c2_b,
               const float* __restrict__ ffn_w, const float* __restrict__ ffn_b,
               const float* __restrict__ alpha, const float* __restrict__ adjA,
               unsigned short* __restrict__ Mws, float* __restrict__ srow, int b0) {
  __shared__ float Pl[V_ * V_];
  __shared__ float xmL[C_ * V_];
  __shared__ float kL[4 * V_], qL[4 * V_];
  __shared__ float aL[8 * V_ * V_];
  __shared__ float sL[V_ * V_];
  int blk = blockIdx.x, h = blk % H_, bl = blk / H_, b = b0 + bl;
  int tid = threadIdx.x;
  for (int i = tid; i < V_ * V_; i += 256) Pl[i] = P[h * V_ * V_ + i];
  __syncthreads();
  // xm = xmean[b] @ P_h
  for (int i = tid; i < C_ * V_; i += 256) {
    int c = i / V_, v = i % V_;
    float s = 0.f;
    const float* xr = xmean + ((size_t)b * C_ + c) * V_;
    for (int u = 0; u < V_; u++) s += xr[u] * Pl[u * V_ + v];
    xmL[i] = s;
  }
  __syncthreads();
  // k, q projections (RC2=4)
  if (tid < 4 * V_) {
    int o = tid / V_, v = tid % V_;
    float s = wk_b[h * 4 + o];
    const float* wr = wk_w + (h * 4 + o) * C_;
    for (int c = 0; c < C_; c++) s += wr[c] * xmL[c * V_ + v];
    kL[tid] = s;
  } else if (tid >= 128 && tid < 128 + 4 * V_) {
    int t2 = tid - 128, o = t2 / V_, v = t2 % V_;
    float s = wq_b[h * 4 + o];
    const float* wr = wq_w + (h * 4 + o) * C_;
    for (int c = 0; c < C_; c++) s += wr[c] * xmL[c * V_ + v];
    qL[t2] = s;
  }
  __syncthreads();
  // a = tanh(concat(c1@ (k x q), c2@ (k - q)))
  for (int i = tid; i < 8 * V_ * V_; i += 256) {
    int cc = i / (V_ * V_), vw = i % (V_ * V_), v = vw / V_, w2 = vw % V_;
    float s;
    if (cc < 4) {
      s = c1_b[h * 4 + cc];
      const float* wr = c1_w + (h * 4 + cc) * 4;
      for (int j = 0; j < 4; j++) s += wr[j] * (kL[j * V_ + v] * qL[j * V_ + w2]);
    } else {
      int c2i = cc - 4;
      s = c2_b[h * 4 + c2i];
      const float* wr = c2_w + (h * 4 + c2i) * 4;
      for (int j = 0; j < 4; j++) s += wr[j] * (kL[j * V_ + v] - qL[j * V_ + w2]);
    }
    aL[i] = tanhf(s);
  }
  __syncthreads();
  float al = alpha[h];
  for (int o = 0; o < CO_; o++) {
    // s[v,w] = alpha*(ffn@a + ffn_b) + adjA
    for (int i = tid; i < V_ * V_; i += 256) {
      int v = i / V_, w2 = i % V_;
      float s = ffn_b[h * CO_ + o];
      const float* wr = ffn_w + (h * CO_ + o) * 8;
      for (int cc = 0; cc < 8; cc++) s += wr[cc] * aL[cc * V_ * V_ + i];
      sL[i] = al * s + adjA[(h * V_ + v) * V_ + w2];
    }
    __syncthreads();
    if (tid < V_) {  // srow[v] = sum_w s[v,w]
      float s = 0.f;
      for (int w2 = 0; w2 < V_; w2++) s += sL[tid * V_ + w2];
      srow[(((size_t)bl * H_ + h) * CO_ + o) * V_ + tid] = s;
    }
    // M[u,v] = sum_w P[u,w]*s[v,w]; store as WMMA B-operand bf16 (zero-padded)
    unsigned short* Mo = Mws + ((((size_t)bl * H_ + h) * CO_ + o) * 2) * 512;
    for (int i = tid; i < 1024; i += 256) {
      int e = i & 15, l = (i >> 4) & 31, vt = i >> 9;
      int v = vt * 16 + (l & 15);
      int u = kmap(e, l);
      float m = 0.f;
      if (u < V_ && v < V_) {
        const float* pr = Pl + u * V_;
        const float* sr = sL + v * V_;
        for (int w2 = 0; w2 < V_; w2++) m += pr[w2] * sr[w2];
      }
      Mo[i] = f2bf(m);
    }
    __syncthreads();
  }
}

// --------------------------- K5: main WMMA kernel ---------------------------
// grid: BCH * 8 (t-tiles) * 4 (o-tiles); 128 threads = 4 waves.
__global__ __launch_bounds__(128)
void k5_main(const float* __restrict__ x, const unsigned short* __restrict__ wvA,
             const unsigned short* __restrict__ Mws, const float* __restrict__ srow,
             const float* __restrict__ wv_b, float* __restrict__ out,
             float* __restrict__ stats, int b0) {
  // Xs: half-C slab of x tile, bf16, ALREADY in WMMA B-operand swizzled layout:
  //   Xs[tile nt][lane][e]  with  K(c_local) = kmap(e, lane), N = nt*16 + lane%16
  __shared__ __align__(16) unsigned short Xs[25 * 512];
  __shared__ __align__(16) unsigned short Yl[16][16][32];  // Y[o][t][u(pad32)] bf16
  __shared__ float ldsS[16], ldsS2[16];
  int blk = blockIdx.x;
  int ot = blk & 3, tt = (blk >> 2) & 7, bl = blk >> 5, b = b0 + bl;
  int tid = threadIdx.x, w = tid >> 5, lane = tid & 31;
  if (tid < 16) { ldsS[tid] = 0.f; ldsS2[tid] = 0.f; }

  v8f acc2[8];  // stage-2 accumulators: 4 o_local x 2 v-tiles, kept across heads
#pragma unroll
  for (int i = 0; i < 8; i++)
#pragma unroll
    for (int e = 0; e < 8; e++) acc2[i][e] = 0.f;

  const int ntc = (28 - w) / 4;  // stage-1 N-tiles owned by this wave (of 25)
  const v16bf* Xv = (const v16bf*)Xs;

  for (int h = 0; h < H_; ++h) {
    __syncthreads();
    {  // zero Y pad with b128 stores
      uint4 z4; z4.x = z4.y = z4.z = z4.w = 0u;
      uint4* Yz = (uint4*)Yl;
      for (int i = tid; i < 1024; i += 128) Yz[i] = z4;
    }
    // prefetch next head's M block for this o-tile (16 KB region)
    if (h + 1 < H_) {
      const unsigned short* mp =
          Mws + ((((size_t)bl * H_ + (h + 1)) * CO_ + ot * 16) * 2) * 512;
      __builtin_prefetch(mp + (size_t)tid * 64, 0, 1);
    }

    v8f acc1[7];
#pragma unroll
    for (int i = 0; i < 7; i++)
#pragma unroll
      for (int e = 0; e < 8; e++) acc1[i][e] = 0.f;

    for (int kk = 0; kk < 2; ++kk) {       // K = C split 2x32
      __syncthreads();
      // Stage x slab -> LDS, swizzling into B-operand layout.
      // Work item i = (octet g of 8 c-rows, j = float4 column). Loads 8
      // coalesced float4 rows, emits 4 packed uint4 (8 bf16) LDS stores.
      for (int i = tid; i < 400; i += 128) {
        int g = i / 100, j = i - g * 100;
        int hi = g & 1, e0 = (g >= 2) ? 8 : 0;
        float4 f[8];
#pragma unroll
        for (int r = 0; r < 8; r++) {
          size_t rowbase = ((size_t)b * C_ + (kk * 32 + g * 8 + r)) * (T_ * V_) + tt * 400;
          f[r] = *(const float4*)(x + rowbase + j * 4);
        }
#pragma unroll
        for (int q = 0; q < 4; q++) {
          int nn = j * 4 + q;
          const float* fp0 = &f[0].x;
          uint4 pk;
          pk.x = pk2((&f[0].x)[q], (&f[1].x)[q]);
          pk.y = pk2((&f[2].x)[q], (&f[3].x)[q]);
          pk.z = pk2((&f[4].x)[q], (&f[5].x)[q]);
          pk.w = pk2((&f[6].x)[q], (&f[7].x)[q]);
          (void)fp0;
          int di = (nn >> 4) * 512 + (hi * 16 + (nn & 15)) * 16 + e0;
          *(uint4*)(Xs + di) = pk;
        }
      }
      __syncthreads();
      const v16bf* Ap = (const v16bf*)(wvA + ((h * 4 + ot) * 2 + kk) * 512);
      v16bf Av = Ap[lane];
#pragma unroll
      for (int it = 0; it < 7; ++it) {
        if (it < ntc) {
          int nt = w + 4 * it;
          v16bf Bv = Xv[nt * 32 + lane];   // 2x ds_load_b128, pre-swizzled
          acc1[it] = __builtin_amdgcn_wmma_f32_16x16x32_bf16(
              false, Av, false, Bv, (short)0, acc1[it], false, false);
        }
      }
    }
    // spill Y tiles (C-layout: m = r + (lane>=16 ? 8 : 0)) to LDS as bf16
#pragma unroll
    for (int it = 0; it < 7; ++it) {
      if (it < ntc) {
        int nt = w + 4 * it;
        int n = nt * 16 + (lane & 15);
        int t = n / V_, u = n - t * V_;
        int mbase = (lane < 16) ? 0 : 8;
#pragma unroll
        for (int r = 0; r < 8; r++) Yl[mbase + r][t][u] = f2bf(acc1[it][r]);
      }
    }
    __syncthreads();
    // stage 2: out_tile[t,v] += Y_o[t,u(32)] @ M_o[u,v]; K=32 in one WMMA
#pragma unroll
    for (int j = 0; j < 4; j++) {
      int ol = w * 4 + j;
      union { unsigned short u[16]; v16bf v; } Au;
      int m = lane & 15;
#pragma unroll
      for (int e = 0; e < 16; e++) Au.u[e] = Yl[ol][m][kmap(e, lane)];
      int o = ot * 16 + ol;
#pragma unroll
      for (int vt = 0; vt < 2; ++vt) {
        const v16bf* Bp = (const v16bf*)(Mws + ((((size_t)bl * H_ + h) * CO_ + o) * 2 + vt) * 512);
        v16bf Bv = Bp[lane];
        acc2[j * 2 + vt] = __builtin_amdgcn_wmma_f32_16x16x32_bf16(
            false, Au.v, false, Bv, (short)0, acc2[j * 2 + vt], false, false);
      }
    }
  }
  // epilogue: + wv_b*srow bias, write pre-BN out, reduce BN stats
#pragma unroll
  for (int j = 0; j < 4; j++) {
    int ol = w * 4 + j, o = ot * 16 + ol;
#pragma unroll
    for (int vt = 0; vt < 2; ++vt) {
      int v = vt * 16 + (lane & 15);
      float sa = 0.f, sb = 0.f;
      if (v < V_) {
        float bias = 0.f;
        for (int h = 0; h < H_; h++)
          bias += wv_b[h * CO_ + o] * srow[(((size_t)bl * H_ + h) * CO_ + o) * V_ + v];
        int mbase = (lane < 16) ? 0 : 8;
#pragma unroll
        for (int r = 0; r < 8; r++) {
          float val = acc2[j * 2 + vt][r] + bias;
          int t = tt * 16 + mbase + r;
          out[(((size_t)b * CO_ + o) * T_ + t) * V_ + v] = val;
          sa += val; sb += val * val;
        }
      }
      atomicAdd(&ldsS[ol], sa);
      atomicAdd(&ldsS2[ol], sb);
    }
  }
  __syncthreads();
  if (tid < 16) {
    atomicAdd(&stats[ot * 16 + tid], ldsS[tid]);
    atomicAdd(&stats[64 + ot * 16 + tid], ldsS2[tid]);
  }
}

// --------------------------- K6: BN stats finalize --------------------------
__global__ void k6_bnstats(const float* __restrict__ stats, const float* __restrict__ bn_w,
                           float* __restrict__ bnp) {
  int o = threadIdx.x;
  if (o < CO_) {
    const float N = (float)B_ * (float)T_ * (float)V_;
    float m = stats[o] / N;
    float var = stats[64 + o] / N - m * m;
    bnp[o] = m;
    bnp[64 + o] = bn_w[o] * rsqrtf(var + 1e-5f);
  }
}

// --------------------------- K7: BN + residual + ReLU -----------------------
__global__ __launch_bounds__(256)
void k7_final(float* __restrict__ out, const float* __restrict__ x,
              const float* __restrict__ bnp, const float* __restrict__ bn_b) {
  size_t n = (size_t)B_ * CO_ * T_ * V_;
  for (size_t i = (size_t)blockIdx.x * 256 + threadIdx.x; i < n; i += (size_t)gridDim.x * 256) {
    int o = (int)((i / (T_ * V_)) % CO_);
    float v = (out[i] - bnp[o]) * bnp[64 + o] + bn_b[o] + x[i];
    out[i] = fmaxf(v, 0.f);
  }
}

// ---------------------------------------------------------------------------
extern "C" void kernel_launch(void* const* d_in, const int* in_sizes, int n_in,
                              void* d_out, int out_size, void* d_ws, size_t ws_size,
                              hipStream_t stream) {
  (void)in_sizes; (void)n_in; (void)out_size; (void)ws_size;
  const float* x         = (const float*)d_in[0];
  const float* log_alpha = (const float*)d_in[1];
  const float* wk_w = (const float*)d_in[2];
  const float* wk_b = (const float*)d_in[3];
  const float* wq_w = (const float*)d_in[4];
  const float* wq_b = (const float*)d_in[5];
  const float* wv_w = (const float*)d_in[6];
  const float* wv_b = (const float*)d_in[7];
  const float* c1_w = (const float*)d_in[8];
  const float* c1_b = (const float*)d_in[9];
  const float* c2_w = (const float*)d_in[10];
  const float* c2_b = (const float*)d_in[11];
  const float* ffn_w = (const float*)d_in[12];
  const float* ffn_b = (const float*)d_in[13];
  const float* alpha = (const float*)d_in[14];
  const float* adjA  = (const float*)d_in[15];
  const float* bn_w  = (const float*)d_in[16];
  const float* bn_b  = (const float*)d_in[17];
  float* out = (float*)d_out;

  char* ws = (char*)d_ws;
  float* P     = (float*)(ws + OFF_P);
  float* xmean = (float*)(ws + OFF_XMEAN);
  float* srow  = (float*)(ws + OFF_SROW);
  float* stats = (float*)(ws + OFF_STATS);
  float* bnp   = (float*)(ws + OFF_BNP);
  unsigned short* wvA = (unsigned short*)(ws + OFF_WVA);
  unsigned short* Mws = (unsigned short*)(ws + OFF_MWS);

  k1_sinkhorn<<<H_, 32, 0, stream>>>(log_alpha, P);
  k2_xmean<<<B_ * C_, 128, 0, stream>>>(x, xmean);
  k0_wva<<<48, 256, 0, stream>>>(wv_w, wvA);
  hipMemsetAsync(stats, 0, 2 * 64 * sizeof(float), stream);

  for (int b0 = 0; b0 < B_; b0 += BCH) {
    k3_scores<<<BCH * H_, 256, 0, stream>>>(xmean, P, wk_w, wk_b, wq_w, wq_b,
                                            c1_w, c1_b, c2_w, c2_b, ffn_w, ffn_b,
                                            alpha, adjA, Mws, srow, b0);
    k5_main<<<BCH * 8 * 4, 128, 0, stream>>>(x, wvA, Mws, srow, wv_b, out, stats, b0);
  }
  k6_bnstats<<<1, 64, 0, stream>>>(stats, bn_w, bnp);
  k7_final<<<4096, 256, 0, stream>>>(out, x, bnp, bn_b);
}
